// OurAdapterLayer_12137577578735
// MI455X (gfx1250) — matrix operand order repaired and checked by reading, not compiled, not run on hardware
//
#include <hip/hip_runtime.h>

// ---------------------------------------------------------------------------
// Problem constants (match reference)
// ---------------------------------------------------------------------------
#define B_ 16
#define T_ 1024
#define D_ 1024
#define H_ 256
#define R_ 64
#define N_ 8

typedef __attribute__((ext_vector_type(16))) __bf16 v16bf;
typedef __attribute__((ext_vector_type(8)))  float  v8f;

union Frag { v16bf v; unsigned u[8]; };

// f32 -> bf16 round-to-nearest-even
__device__ __forceinline__ unsigned short f2bf(float f) {
    unsigned u = __float_as_uint(f);
    u += 0x7FFFu + ((u >> 16) & 1u);
    return (unsigned short)(u >> 16);
}

// ---------------------------------------------------------------------------
// WMMA helper (CDNA5 16x16x32 bf16, f32 accumulate)
// ---------------------------------------------------------------------------
__device__ __forceinline__ v8f wmma_bf16(v16bf a, v16bf b, v8f c) {
    return __builtin_amdgcn_wmma_f32_16x16x32_bf16(
        /*neg_a=*/false, a, /*neg_b=*/false, b,
        /*c_mod=*/(short)0, c, /*reuse_a=*/false, /*reuse_b=*/false);
}

// A fragment: 16x32 bf16 tile from LDS (row-major, leading dim ldk).
// Lanes 0-15 hold K {0-7,16-23}, lanes 16-31 hold {8-15,24-31}; row M = lane&15.
// Pairs of consecutive K share a dword -> 8 dword loads (merged to b128 by LLVM).
__device__ __forceinline__ v16bf load_a(const unsigned short* lds, int row,
                                        int ldk, int lane) {
    Frag f;
    const int half8 = (lane >> 4) << 3;
    const unsigned short* p = lds + row * ldk;
#pragma unroll
    for (int v = 0; v < 8; ++v) {
        int j = v * 2;
        int k = ((j & 8) << 1) + (j & 7) + half8;   // {0,2,4,6,16,18,20,22}+half*8
        f.u[v] = *(const unsigned*)(p + k);
    }
    return f.v;
}

// B fragment: 32x16 bf16 tile.  LDS chunk is column-major: lds[n*32 + k].
// Lane holds column n = lane&15; lanes 0-15 cover K {0-7,16-23}, lanes 16-31 K {8-15,24-31}.
__device__ __forceinline__ v16bf load_b(const unsigned short* ldsB, int nLocal,
                                        int lane) {
    Frag f;
    const unsigned short* p = ldsB + nLocal * 32 + ((lane >> 4) << 3);
#pragma unroll
    for (int v = 0; v < 8; ++v) f.u[v] = *(const unsigned*)(p + ((v & 4) << 2) + 2 * (v & 3));
    return f.v;
}

// ---------------------------------------------------------------------------
// Staging: global f32 -> LDS bf16 (256-thread block)
// ---------------------------------------------------------------------------
// x chunk: 64 rows x 32 K, source row-major with leading dim ld
__device__ __forceinline__ void stage_a(unsigned short* lds, const float* src,
                                        int ld, int k0, int tid) {
    int r  = tid >> 2;
    int c0 = (tid & 3) * 8;
    const float* s = src + (size_t)r * ld + k0 + c0;
    float4 v0 = *(const float4*)(s);
    float4 v1 = *(const float4*)(s + 4);
    unsigned short* d = lds + r * 32 + c0;
    d[0] = f2bf(v0.x); d[1] = f2bf(v0.y); d[2] = f2bf(v0.z); d[3] = f2bf(v0.w);
    d[4] = f2bf(v1.x); d[5] = f2bf(v1.y); d[6] = f2bf(v1.z); d[7] = f2bf(v1.w);
}

// weight chunk: 32 K x NCOLS columns from row-major [K x ldw]; stored col-major.
// Also prefetches the chunk-after-next to keep the L2->WGP path warm.
template <int NCOLS>
__device__ __forceinline__ void stage_b(unsigned short* lds, const float* src,
                                        int ldw, int k0, int col0, int tid) {
    constexpr int TPC = 256 / NCOLS;   // threads per column
    constexpr int KP  = 32 / TPC;      // K elements per thread
    int n  = tid % NCOLS;
    int ks = (tid / NCOLS) * KP;
#pragma unroll
    for (int i = 0; i < KP; ++i) {
        int k = ks + i;
        lds[n * 32 + k] = f2bf(src[(size_t)(k0 + k) * ldw + col0 + n]);
    }
    __builtin_prefetch(src + (size_t)(k0 + 64 + ks) * ldw + col0 + n, 0, 1);
}

// ---------------------------------------------------------------------------
// Kernel 0: zero the loss accumulators
// ---------------------------------------------------------------------------
__global__ void init_losses_kernel(float* losses) {
    losses[threadIdx.x] = 0.0f;
}

// ---------------------------------------------------------------------------
// Kernel 1: per-(n,b) autoencoder reconstruction loss, WMMA bf16
//   grid (T/64, B, N), 256 threads, double-buffered staging
// ---------------------------------------------------------------------------
__global__ __launch_bounds__(256)
void disc_loss_kernel(const float* __restrict__ x,
                      const float* __restrict__ W_enc,
                      const float* __restrict__ b_enc,
                      const float* __restrict__ W_dec,
                      const float* __restrict__ b_dec,
                      float* __restrict__ losses) {
    __shared__ unsigned short ldsA[2][64 * 32];    //  8 KB  x chunks (dbl buf)
    __shared__ unsigned short ldsB[2][128 * 32];   // 16 KB  weight chunks (dbl buf)
    __shared__ unsigned short ldsH[64 * 256];      // 32 KB  h tile (bf16)
    __shared__ float red[256];

    const int tid  = threadIdx.x;
    const int lane = tid & 31;
    const int w    = tid >> 5;
    const int n    = blockIdx.z;
    const int bb   = blockIdx.y;
    const int t0   = blockIdx.x * 64;

    const float* xt = x + ((size_t)bb * T_ + t0) * D_;
    const float* We = W_enc + (size_t)n * D_ * H_;
    const float* Wd = W_dec + (size_t)n * H_ * D_;

    const int rowBase = (w & 3) * 16;      // 4 row tiles of 16
    const int colGrp  = w >> 2;            // 2 column groups per 128-col block
    const int aRow    = rowBase + (lane & 15);
    const int half8   = (lane >> 4) << 3;

    // ---- Phase 1: h = relu(x @ W_enc + b_enc)  ->  ldsH[64][256] (bf16)
    for (int cb = 0; cb < 2; ++cb) {               // 2 blocks of 128 H-columns
        v8f acc[4];
#pragma unroll
        for (int i = 0; i < 4; ++i) acc[i] = (v8f){0,0,0,0,0,0,0,0};

        stage_a(ldsA[0], xt, D_, 0, tid);
        stage_b<128>(ldsB[0], We, H_, 0, cb * 128, tid);
        __syncthreads();
        for (int ks = 0; ks < 32; ++ks) {          // K = 1024 in steps of 32
            const int cur = ks & 1;
            if (ks + 1 < 32) {
                stage_a(ldsA[cur ^ 1], xt, D_, (ks + 1) * 32, tid);
                stage_b<128>(ldsB[cur ^ 1], We, H_, (ks + 1) * 32, cb * 128, tid);
            }
            v16bf af = load_a(ldsA[cur], aRow, 32, lane);
#pragma unroll
            for (int i = 0; i < 4; ++i) {
                v16bf bf = load_b(ldsB[cur], (colGrp * 4 + i) * 16 + (lane & 15), lane);
                acc[i] = wmma_bf16(af, bf, acc[i]);
            }
            __syncthreads();                        // one barrier per K-step
        }
#pragma unroll
        for (int i = 0; i < 4; ++i) {
            int col = cb * 128 + (colGrp * 4 + i) * 16 + (lane & 15);
            float be = b_enc[n * H_ + col];
#pragma unroll
            for (int e = 0; e < 8; ++e) {
                int row = rowBase + e + half8;
                float v = acc[i][e] + be;
                v = v > 0.0f ? v : 0.0f;
                ldsH[row * 256 + col] = f2bf(v);
            }
        }
        __syncthreads();
    }

    // ---- Phase 2: recon = h @ W_dec + b_dec ; err += (recon - x)^2
    float err = 0.0f;
    for (int cb = 0; cb < 8; ++cb) {               // 8 blocks of 128 D-columns
        v8f acc[4];
#pragma unroll
        for (int i = 0; i < 4; ++i) acc[i] = (v8f){0,0,0,0,0,0,0,0};

        stage_b<128>(ldsB[0], Wd, D_, 0, cb * 128, tid);
        __syncthreads();
        for (int ks = 0; ks < 8; ++ks) {           // K = 256 in steps of 32
            const int cur = ks & 1;
            if (ks + 1 < 8)
                stage_b<128>(ldsB[cur ^ 1], Wd, D_, (ks + 1) * 32, cb * 128, tid);
            v16bf af = load_a(ldsH + ks * 32, aRow, 256, lane);
#pragma unroll
            for (int i = 0; i < 4; ++i) {
                v16bf bf = load_b(ldsB[cur], (colGrp * 4 + i) * 16 + (lane & 15), lane);
                acc[i] = wmma_bf16(af, bf, acc[i]);
            }
            __syncthreads();
        }
#pragma unroll
        for (int i = 0; i < 4; ++i) {
            int col = cb * 128 + (colGrp * 4 + i) * 16 + (lane & 15);
            float bd = b_dec[n * D_ + col];
#pragma unroll
            for (int e = 0; e < 8; ++e) {
                int row = rowBase + e + half8;
                float v  = acc[i][e] + bd;
                float xv = xt[(size_t)row * D_ + col];   // hits L2 (x fits)
                float d  = v - xv;
                err += d * d;
            }
        }
    }

    red[tid] = err;
    __syncthreads();
    for (int s = 128; s > 0; s >>= 1) {
        if (tid < s) red[tid] += red[tid + s];
        __syncthreads();
    }
    if (tid == 0) atomicAdd(losses + n * B_ + bb, red[0]);
}

// ---------------------------------------------------------------------------
// Kernel 2: per-sample argmin routing
// ---------------------------------------------------------------------------
__global__ void route_kernel(const float* __restrict__ losses,
                             const int* __restrict__ cidx,
                             int* __restrict__ adp) {
    int b = threadIdx.x;
    if (b >= B_) return;
    float best = losses[b];
    int bi = 0;
#pragma unroll
    for (int n = 1; n < N_; ++n) {
        float v = losses[n * B_ + b];
        if (v < best) { best = v; bi = n; }
    }
    adp[b] = cidx[bi];
}

// ---------------------------------------------------------------------------
// Kernel 3: out = x@W_base + b_base + relu(x@W_down[a]+b_down[a])@W_up[a] + b_up[a]
//   grid (T/64, B), 256 threads, dynamic LDS (x tile resident in bf16),
//   double-buffered weight staging
// ---------------------------------------------------------------------------
__global__ __launch_bounds__(256)
void fused_out_kernel(const float* __restrict__ x,
                      const float* __restrict__ W_base,
                      const float* __restrict__ b_base,
                      const float* __restrict__ W_down,
                      const float* __restrict__ b_down,
                      const float* __restrict__ W_up,
                      const float* __restrict__ b_up,
                      const int* __restrict__ adp,
                      float* __restrict__ out) {
    extern __shared__ unsigned short smem[];
    unsigned short* ldsX  = smem;                    // 64*1024 bf16 = 128 KB
    unsigned short* ldsB0 = ldsX + 64 * 1024;        // 128*32  bf16 =   8 KB
    unsigned short* ldsB1 = ldsB0 + 128 * 32;        // 128*32  bf16 =   8 KB
    unsigned short* ldsM  = ldsB1 + 128 * 32;        // 64*64   bf16 =   8 KB
    unsigned short* ldsBB[2] = { ldsB0, ldsB1 };

    const int tid  = threadIdx.x;
    const int lane = tid & 31;
    const int w    = tid >> 5;
    const int bb   = blockIdx.y;
    const int t0   = blockIdx.x * 64;
    const int a    = adp[bb];

    const float* xt  = x + ((size_t)bb * T_ + t0) * D_;
    const float* Wdn = W_down + (size_t)a * D_ * R_;
    const float* Wup = W_up   + (size_t)a * R_ * D_;

    // Stage full 64x1024 x tile as bf16 (one quarter-row per thread)
    {
        int r  = tid >> 2;
        int c0 = (tid & 3) * 256;
        const float* s = xt + (size_t)r * D_ + c0;
        unsigned short* d = ldsX + r * 1024 + c0;
        for (int i = 0; i < 256; i += 4) {
            float4 v = *(const float4*)(s + i);
            d[i + 0] = f2bf(v.x); d[i + 1] = f2bf(v.y);
            d[i + 2] = f2bf(v.z); d[i + 3] = f2bf(v.w);
        }
    }
    __syncthreads();

    const int rowBase = (w & 3) * 16;
    const int colGrp  = w >> 2;
    const int aRow    = rowBase + (lane & 15);
    const int half8   = (lane >> 4) << 3;

    // ---- mid = relu(x @ W_down[a] + b_down[a])   (64 x 64) -> ldsM
    {
        v8f acc[2];
#pragma unroll
        for (int i = 0; i < 2; ++i) acc[i] = (v8f){0,0,0,0,0,0,0,0};

        stage_b<64>(ldsBB[0], Wdn, R_, 0, 0, tid);
        __syncthreads();
        for (int ks = 0; ks < 32; ++ks) {          // K = 1024
            const int cur = ks & 1;
            if (ks + 1 < 32)
                stage_b<64>(ldsBB[cur ^ 1], Wdn, R_, (ks + 1) * 32, 0, tid);
            v16bf af = load_a(ldsX + ks * 32, aRow, 1024, lane);
#pragma unroll
            for (int i = 0; i < 2; ++i) {
                v16bf bf = load_b(ldsBB[cur], (colGrp * 2 + i) * 16 + (lane & 15), lane);
                acc[i] = wmma_bf16(af, bf, acc[i]);
            }
            __syncthreads();
        }
#pragma unroll
        for (int i = 0; i < 2; ++i) {
            int col = (colGrp * 2 + i) * 16 + (lane & 15);
            float bd = b_down[a * R_ + col];
#pragma unroll
            for (int e = 0; e < 8; ++e) {
                int row = rowBase + e + half8;
                float v = acc[i][e] + bd;
                v = v > 0.0f ? v : 0.0f;
                ldsM[row * 64 + col] = f2bf(v);
            }
        }
        __syncthreads();
    }

    // ---- out = x@W_base + mid@W_up + b_base + b_up  (both into same WMMA C)
    for (int cb = 0; cb < 8; ++cb) {               // 8 blocks of 128 D-columns
        v8f acc[4];
#pragma unroll
        for (int i = 0; i < 4; ++i) acc[i] = (v8f){0,0,0,0,0,0,0,0};

        stage_b<128>(ldsBB[0], W_base, D_, 0, cb * 128, tid);
        __syncthreads();
        for (int ks = 0; ks < 32; ++ks) {          // base GEMM, K = 1024
            const int cur = ks & 1;
            if (ks + 1 < 32)
                stage_b<128>(ldsBB[cur ^ 1], W_base, D_, (ks + 1) * 32, cb * 128, tid);
            else
                stage_b<128>(ldsBB[cur ^ 1], Wup, D_, 0, cb * 128, tid);  // head of up GEMM
            v16bf af = load_a(ldsX + ks * 32, aRow, 1024, lane);
#pragma unroll
            for (int i = 0; i < 4; ++i) {
                v16bf bf = load_b(ldsBB[cur], (colGrp * 4 + i) * 16 + (lane & 15), lane);
                acc[i] = wmma_bf16(af, bf, acc[i]);
            }
            __syncthreads();
        }
        for (int ks = 0; ks < 2; ++ks) {           // up GEMM, K = 64, same accumulators
            const int cur = ks & 1;                // buffer 0 pre-staged above
            if (ks + 1 < 2)
                stage_b<128>(ldsBB[cur ^ 1], Wup, D_, (ks + 1) * 32, cb * 128, tid);
            v16bf af = load_a(ldsM + ks * 32, aRow, 64, lane);
#pragma unroll
            for (int i = 0; i < 4; ++i) {
                v16bf bf = load_b(ldsBB[cur], (colGrp * 4 + i) * 16 + (lane & 15), lane);
                acc[i] = wmma_bf16(af, bf, acc[i]);
            }
            __syncthreads();
        }
#pragma unroll
        for (int i = 0; i < 4; ++i) {
            int col = cb * 128 + (colGrp * 4 + i) * 16 + (lane & 15);
            float bias = b_base[col] + b_up[a * D_ + col];
#pragma unroll
            for (int e = 0; e < 8; ++e) {
                int row = rowBase + e + half8;
                out[((size_t)bb * T_ + t0 + row) * D_ + col] = acc[i][e] + bias;
            }
        }
    }
}

// ---------------------------------------------------------------------------
// Host entry
// ---------------------------------------------------------------------------
extern "C" void kernel_launch(void* const* d_in, const int* in_sizes, int n_in,
                              void* d_out, int out_size, void* d_ws, size_t ws_size,
                              hipStream_t stream) {
    (void)in_sizes; (void)n_in; (void)out_size; (void)ws_size;

    const float* x      = (const float*)d_in[0];
    const float* W_base = (const float*)d_in[1];
    const float* b_base = (const float*)d_in[2];
    const float* W_enc  = (const float*)d_in[3];
    const float* b_enc  = (const float*)d_in[4];
    const float* W_dec  = (const float*)d_in[5];
    const float* b_dec  = (const float*)d_in[6];
    const float* W_down = (const float*)d_in[7];
    const float* b_down = (const float*)d_in[8];
    const float* W_up   = (const float*)d_in[9];
    const float* b_up   = (const float*)d_in[10];
    const int*   cidx   = (const int*)d_in[11];

    float* losses = (float*)d_ws;                       // N*B floats
    int*   adp    = (int*)((char*)d_ws + 512);          // B ints
    float* out    = (float*)d_out;

    hipLaunchKernelGGL(init_losses_kernel, dim3(1), dim3(N_ * B_), 0, stream, losses);

    hipLaunchKernelGGL(disc_loss_kernel, dim3(T_ / 64, B_, N_), dim3(256), 0, stream,
                       x, W_enc, b_enc, W_dec, b_dec, losses);

    hipLaunchKernelGGL(route_kernel, dim3(1), dim3(B_), 0, stream, losses, cidx, adp);

    const size_t smem = (size_t)(64 * 1024 + 2 * 128 * 32 + 64 * 64) * sizeof(unsigned short);
    hipLaunchKernelGGL(fused_out_kernel, dim3(T_ / 64, B_), dim3(256), smem, stream,
                       x, W_base, b_base, W_down, b_down, W_up, b_up, adp, out);
}
